// MultiHeadAttention_10608569221303
// MI455X (gfx1250) — matrix operand order repaired and checked by reading, not compiled, not run on hardware
//
#include <hip/hip_runtime.h>

typedef _Float16 f16;
typedef __attribute__((ext_vector_type(16))) _Float16 v16h;
typedef __attribute__((ext_vector_type(8)))  _Float16 v8h;
typedef __attribute__((ext_vector_type(4)))  _Float16 v4h;
typedef __attribute__((ext_vector_type(8)))  float    v8f;

#define B_   2
#define S_   2048
#define E_   1024
#define H_   16
#define D_   64
#define AHS_ 1024

union ABu { v16h v; v8h h[2]; };

static __device__ __forceinline__ v8f wmma16(v16h a, v16h b, v8f c) {
  // D = A(16x32 f16) * B(32x16 f16) + C(16x16 f32)
  return __builtin_amdgcn_wmma_f32_16x16x32_f16(false, a, false, b, (short)0, c,
                                                false, false);
}

// ---------------------------------------------------------------- f32 -> f16
__global__ __launch_bounds__(256) void cvt_f16_kernel(const float* __restrict__ src,
                                                      f16* __restrict__ dst, int n4) {
  int i = blockIdx.x * 256 + threadIdx.x;
  if (i < n4) {
    float4 f = ((const float4*)src)[i];
    v4h o;
    o[0] = (f16)f.x; o[1] = (f16)f.y; o[2] = (f16)f.z; o[3] = (f16)f.w;
    ((v4h*)dst)[i] = o;
  }
}

// ------------------------------------------------- Q/K/V projection (WMMA GEMM)
// grid = (M/64, N/64, 3)   block = 128 (4 waves), wave w -> rows 16w..16w+15
// Y = X[4096,1024] @ W^T ; z=0 -> Q (scaled 1/8, [b,h,s,d])
//                          z=1 -> K ([b,h,s,d]) ; z=2 -> V^T ([b,h,d,s])
__global__ __launch_bounds__(128) void proj_kernel(const f16* __restrict__ X,
                                                   const f16* __restrict__ Wq,
                                                   const f16* __restrict__ Wk,
                                                   const f16* __restrict__ Wv,
                                                   f16* __restrict__ Q,
                                                   f16* __restrict__ K,
                                                   f16* __restrict__ Vt) {
  const int z    = blockIdx.z;
  const f16* W   = (z == 0) ? Wq : ((z == 1) ? Wk : Wv);
  const int wave = threadIdx.x >> 5;
  const int lane = threadIdx.x & 31;
  const int r    = lane & 15;     // row (A) / col (B,C) selector
  const int half = lane >> 4;

  const int m0 = blockIdx.x * 64 + wave * 16;
  const int n0 = blockIdx.y * 64;

  const v8f vz = {0.f, 0.f, 0.f, 0.f, 0.f, 0.f, 0.f, 0.f};
  v8f acc[4] = {vz, vz, vz, vz};

  const f16* arow = X + (size_t)(m0 + r) * E_;
  for (int k = 0; k < E_; k += 32) {
    ABu A;
    const f16* ap = arow + k + half * 8;
    A.h[0] = *(const v8h*)ap;
    A.h[1] = *(const v8h*)(ap + 16);
#pragma unroll
    for (int nt = 0; nt < 4; ++nt) {
      const f16* bp = W + (size_t)(n0 + nt * 16 + r) * E_ + k + half * 8;
      ABu Bu; Bu.h[0] = *(const v8h*)bp; Bu.h[1] = *(const v8h*)(bp + 16);
      v16h Bv = Bu.v;
      acc[nt] = wmma16(A.v, Bv, acc[nt]);
    }
  }

  const float qscale = 0.125f;  // 1/sqrt(D), folded into Q
#pragma unroll
  for (int nt = 0; nt < 4; ++nt) {
#pragma unroll
    for (int g = 0; g < 8; ++g) {
      float val = acc[nt][g];
      int grow = m0 + g + 8 * half;        // global token row
      int b    = grow >> 11;               // /S_
      int s    = grow & (S_ - 1);
      int col  = n0 + nt * 16 + r;         // output channel a = h*64+d
      int h    = col >> 6;
      int d    = col & 63;
      if (z == 0) {
        Q[(((size_t)(b * H_ + h)) * S_ + s) * D_ + d] = (f16)(val * qscale);
      } else if (z == 1) {
        K[(((size_t)(b * H_ + h)) * S_ + s) * D_ + d] = (f16)val;
      } else {
        Vt[(((size_t)(b * H_ + h)) * D_ + d) * S_ + s] = (f16)val;
      }
    }
  }
}

// ------------------------------------------------------- fused flash attention
// grid = (S/64, B*H), block = 128 (4 waves); wave owns 16 query rows.
// Online softmax over key tiles of 64; P bounced via wave-private LDS tile.
__global__ __launch_bounds__(128) void attn_kernel(const f16* __restrict__ Q,
                                                   const f16* __restrict__ K,
                                                   const f16* __restrict__ Vt,
                                                   float* __restrict__ out,
                                                   float* __restrict__ ctx) {
  __shared__ __align__(128) f16 lds[4 * 16 * 72];  // 72 = 64 + 8 pad (16B rows)

  const int wave = threadIdx.x >> 5;
  const int lane = threadIdx.x & 31;
  const int r    = lane & 15;
  const int half = lane >> 4;

  const int bh = blockIdx.y;                 // b*H + h
  const int q0 = blockIdx.x * 64 + wave * 16;

  const f16* Qb = Q  + ((size_t)bh * S_ + q0) * D_;
  const f16* Kb = K  + (size_t)bh * S_ * D_;
  const f16* Vb = Vt + (size_t)bh * D_ * S_;
  f16* myLds = lds + wave * 16 * 72;

  // Q A-operand (kept in registers for the whole key loop)
  ABu Aq[2];
#pragma unroll
  for (int kk = 0; kk < 2; ++kk) {
    const f16* qp = Qb + (size_t)r * D_ + kk * 32 + half * 8;
    Aq[kk].h[0] = *(const v8h*)qp;
    Aq[kk].h[1] = *(const v8h*)(qp + 16);
  }

  const v8f vz = {0.f, 0.f, 0.f, 0.f, 0.f, 0.f, 0.f, 0.f};
  v8f acc[4] = {vz, vz, vz, vz};
  float m_i[8], l_i[8];
#pragma unroll
  for (int g = 0; g < 8; ++g) { m_i[g] = -1.0e30f; l_i[g] = 0.f; }

  for (int j0 = 0; j0 < S_; j0 += 64) {
    // ---- scores: S = Q * K^T   (16 x 64, f32)
    v8f sc[4] = {vz, vz, vz, vz};
#pragma unroll
    for (int jt = 0; jt < 4; ++jt) {
#pragma unroll
      for (int kk = 0; kk < 2; ++kk) {
        const f16* kp = Kb + (size_t)(j0 + jt * 16 + r) * D_ + kk * 32 + half * 8;
        ABu Bu; Bu.h[0] = *(const v8h*)kp; Bu.h[1] = *(const v8h*)(kp + 16);
        v16h Bv = Bu.v;
        sc[jt] = wmma16(Aq[kk].v, Bv, sc[jt]);
      }
    }

    // ---- online softmax (rows g live in lanes of this half only)
    float scale[8];
#pragma unroll
    for (int g = 0; g < 8; ++g) {
      float t = fmaxf(fmaxf(sc[0][g], sc[1][g]), fmaxf(sc[2][g], sc[3][g]));
#pragma unroll
      for (int off = 8; off >= 1; off >>= 1)
        t = fmaxf(t, __shfl_xor(t, off, 32));
      float mnew = fmaxf(m_i[g], t);
      scale[g]   = __expf(m_i[g] - mnew);
      m_i[g]     = mnew;
    }

    float rs[8];
#pragma unroll
    for (int g = 0; g < 8; ++g) {
      float p0 = __expf(sc[0][g] - m_i[g]);
      float p1 = __expf(sc[1][g] - m_i[g]);
      float p2 = __expf(sc[2][g] - m_i[g]);
      float p3 = __expf(sc[3][g] - m_i[g]);
      rs[g] = p0 + p1 + p2 + p3;
      int row = g + 8 * half;
      myLds[row * 72 +  0 + r] = (f16)p0;
      myLds[row * 72 + 16 + r] = (f16)p1;
      myLds[row * 72 + 32 + r] = (f16)p2;
      myLds[row * 72 + 48 + r] = (f16)p3;
    }

#pragma unroll
    for (int g = 0; g < 8; ++g) {
      float t = rs[g];
#pragma unroll
      for (int off = 8; off >= 1; off >>= 1)
        t += __shfl_xor(t, off, 32);
      l_i[g] = l_i[g] * scale[g] + t;
    }

#pragma unroll
    for (int dt = 0; dt < 4; ++dt)
#pragma unroll
      for (int g = 0; g < 8; ++g)
        acc[dt][g] *= scale[g];

    asm volatile("s_wait_dscnt 0x0" ::: "memory");
    __builtin_amdgcn_wave_barrier();

    // ---- ctx += P * V   (A = P from LDS in A-layout, B = rows of V^T)
#pragma unroll
    for (int kk = 0; kk < 2; ++kk) {
      ABu Ap;
      const f16* pp = myLds + r * 72 + kk * 32 + half * 8;
      Ap.h[0] = *(const v8h*)pp;
      Ap.h[1] = *(const v8h*)(pp + 16);
#pragma unroll
      for (int dt = 0; dt < 4; ++dt) {
        const f16* vp = Vb + (size_t)(dt * 16 + r) * S_ + j0 + kk * 32 + half * 8;
        ABu Bu; Bu.h[0] = *(const v8h*)vp; Bu.h[1] = *(const v8h*)(vp + 16);
        v16h Bv = Bu.v;
        acc[dt] = wmma16(Ap.v, Bv, acc[dt]);
      }
    }
    __builtin_amdgcn_wave_barrier();
  }

  // ---- epilogue: normalize, write out[b,s,h*64+d] and ctx[b,h,s,d]
  float inv[8];
#pragma unroll
  for (int g = 0; g < 8; ++g) inv[g] = 1.0f / l_i[g];

  const int b  = bh >> 4;
  const int hh = bh & 15;
#pragma unroll
  for (int dt = 0; dt < 4; ++dt) {
#pragma unroll
    for (int g = 0; g < 8; ++g) {
      float val = acc[dt][g] * inv[g];
      int qrow  = q0 + g + 8 * half;
      int dcol  = dt * 16 + r;
      out[((size_t)b * S_ + qrow) * AHS_ + hh * D_ + dcol] = val;
      ctx[((size_t)bh * S_ + qrow) * D_ + dcol]            = val;
    }
  }
}

// ------------------------------------------------------------------- launcher
extern "C" void kernel_launch(void* const* d_in, const int* in_sizes, int n_in,
                              void* d_out, int out_size, void* d_ws, size_t ws_size,
                              hipStream_t stream) {
  const float* x  = (const float*)d_in[0];
  const float* Wq = (const float*)d_in[1];
  const float* Wk = (const float*)d_in[2];
  const float* Wv = (const float*)d_in[3];

  char* ws = (char*)d_ws;                 // 38 MB used
  f16* Xh  = (f16*)(ws);                  // 8 MB  : [4096,1024]
  f16* Wqh = (f16*)(ws + (8ull  << 20));  // 2 MB
  f16* Wkh = (f16*)(ws + (10ull << 20));  // 2 MB
  f16* Wvh = (f16*)(ws + (12ull << 20));  // 2 MB
  f16* Qh  = (f16*)(ws + (14ull << 20));  // 8 MB  : [b,h,s,d], pre-scaled
  f16* Kh  = (f16*)(ws + (22ull << 20));  // 8 MB  : [b,h,s,d]
  f16* Vth = (f16*)(ws + (30ull << 20));  // 8 MB  : [b,h,d,s]

  const int nx = B_ * S_ * E_;   // 4194304
  const int nw = AHS_ * E_;      // 1048576
  cvt_f16_kernel<<<nx / 4 / 256, 256, 0, stream>>>(x,  Xh,  nx / 4);
  cvt_f16_kernel<<<nw / 4 / 256, 256, 0, stream>>>(Wq, Wqh, nw / 4);
  cvt_f16_kernel<<<nw / 4 / 256, 256, 0, stream>>>(Wk, Wkh, nw / 4);
  cvt_f16_kernel<<<nw / 4 / 256, 256, 0, stream>>>(Wv, Wvh, nw / 4);

  dim3 gp(B_ * S_ / 64, AHS_ / 64, 3);
  proj_kernel<<<gp, 128, 0, stream>>>(Xh, Wqh, Wkh, Wvh, Qh, Kh, Vth);

  float* outp = (float*)d_out;
  float* ctxp = outp + (size_t)B_ * S_ * AHS_;
  dim3 ga(S_ / 64, B_ * H_);
  attn_kernel<<<ga, 128, 0, stream>>>(Qh, Kh, Vth, outp, ctxp);
}
